// SlidingWindowAttention_72215580115482
// MI455X (gfx1250) — matrix operand order repaired and checked by reading, not compiled, not run on hardware
//
#include <hip/hip_runtime.h>
#include <math.h>

#define EMBED 512
#define BATCH 4
#define SEQ   2048
#define WIN   16
#define PADLEN (SEQ + 2 * WIN)          // 2080
#define WINDOW (2 * WIN + 1)            // 33
#define ROWS_TOTAL (BATCH * PADLEN)     // 8320
#define OUT_ROWS   (BATCH * SEQ)        // 8192
#define LDA 520                         // bf16 elems per LDS row (pad vs 512 to stagger banks)
#define TS  32                          // queries per block in the attention kernel
#define ROWS_SH (TS + WINDOW - 1)       // 64 shared Vp rows per block

typedef __attribute__((ext_vector_type(16))) __bf16 v16bf;
typedef __attribute__((ext_vector_type(8)))  __bf16 v8bf;
typedef __attribute__((ext_vector_type(8)))  float  v8f;

// ---- CDNA5 async global->LDS path (guarded; falls back to plain copy) ------------------
#if defined(__has_builtin)
#if __has_builtin(__builtin_amdgcn_global_load_async_to_lds_b128)
#define HAVE_ASYNC_LDS 1
#endif
#endif

typedef int v4i __attribute__((vector_size(16)));             // matches builtin param type
typedef __attribute__((address_space(1))) v4i as1_v4i;        // global
typedef __attribute__((address_space(3))) v4i as3_v4i;        // LDS

__device__ __forceinline__ void copy16_to_lds(const float* gsrc, float* lds) {
#ifdef HAVE_ASYNC_LDS
  __builtin_amdgcn_global_load_async_to_lds_b128((as1_v4i*)gsrc, (as3_v4i*)lds, 0, 0);
#else
  *(float4*)lds = *(const float4*)gsrc;
#endif
}

__device__ __forceinline__ void copy_join() {
#ifdef HAVE_ASYNC_LDS
#if __has_builtin(__builtin_amdgcn_s_wait_asynccnt)
  __builtin_amdgcn_s_wait_asynccnt(0);
#endif
#endif
  __syncthreads();
}

__device__ __forceinline__ v8f wmma_bf16(v16bf a, v16bf b, v8f c) {
  // D = A(16x32 bf16) * B(32x16 bf16) + C(16x16 f32)
  return __builtin_amdgcn_wmma_f32_16x16x32_bf16(false, a, false, b, (short)0, c,
                                                 false, false);
}

__device__ __forceinline__ void cvt_split(float v, __bf16& h, __bf16& l) {
  h = (__bf16)v;
  l = (__bf16)(v - (float)h);
}

// A fragment (16x32 bf16) in contiguous k order, same as load_bfrag:
// lane L<16: M=L, elems 0..15 = K[kb..kb+15]
// lane L>=16: M=L-16, elems 0..15 = K[kb+16..kb+31]
__device__ __forceinline__ v16bf load_afrag(const __bf16* As, int mbase, int kb, int lane) {
  const int m = mbase + (lane & 15);
  const int k = kb + ((lane >> 4) << 4);
  const __bf16* p = As + m * LDA + k;          // 16B aligned: LDA*2=1040=65*16, k mult of 8
  v8bf x0 = *(const v8bf*)p;
  v8bf x1 = *(const v8bf*)(p + 8);
  v16bf a;
#pragma unroll
  for (int j = 0; j < 8; ++j) { a[j] = x0[j]; a[8 + j] = x1[j]; }
  return a;
}

// B fragment (32x16 bf16), B[k][n] = W[n][k], from a PRE-SPLIT bf16 plane [n][k]:
// lane L<16: N=L, K = kb..kb+15; lane L>=16: N=L-16, K = kb+16..kb+31.
__device__ __forceinline__ v16bf load_bfrag(const __bf16* __restrict__ Wb, int n0, int kb,
                                            int lane) {
  const int n = n0 + (lane & 15);
  const int k = kb + ((lane >> 4) << 4);
  return *(const v16bf*)(Wb + (size_t)n * EMBED + k);
}

// ---------------- Kernel 0: one-time fp32 -> bf16 hi/lo split of a weight matrix --------
__global__ __launch_bounds__(256) void wsplit_kernel(const float* __restrict__ W,
                                                     __bf16* __restrict__ hi,
                                                     __bf16* __restrict__ lo) {
  const int i = (blockIdx.x * 256 + threadIdx.x) * 4;
  float4 v = *(const float4*)(W + i);
  __bf16 h, l;
  cvt_split(v.x, h, l); hi[i + 0] = h; lo[i + 0] = l;
  cvt_split(v.y, h, l); hi[i + 1] = h; lo[i + 1] = l;
  cvt_split(v.z, h, l); hi[i + 2] = h; lo[i + 2] = l;
  cvt_split(v.w, h, l); hi[i + 3] = h; lo[i + 3] = l;
}

// ---------------- Kernel 1: fused QKV GEMM -> e (per-row Q.K dot) and Vp ----------------
__global__ __launch_bounds__(256) void qkv_e_v_kernel(
    const float* __restrict__ x,
    const __bf16* __restrict__ Wqh, const __bf16* __restrict__ Wql, const float* __restrict__ bq,
    const __bf16* __restrict__ Wkh, const __bf16* __restrict__ Wkl, const float* __restrict__ bk,
    const __bf16* __restrict__ Wvh, const __bf16* __restrict__ Wvl, const float* __restrict__ bv,
    float* __restrict__ eOut, float* __restrict__ Vp) {
  extern __shared__ __bf16 smem[];
  __bf16* As_hi = smem;                 // [64][LDA]
  __bf16* As_lo = smem + 64 * LDA;      // [64][LDA]
  __shared__ float eSh[64];

  const int tid = threadIdx.x;
  const int strip = blockIdx.x * 64;    // padded-row strip

  if (tid < 64) eSh[tid] = 0.0f;

  // Stage x strip into LDS as split bf16 (zero outside the valid sequence = padding rows)
  for (int idx = tid; idx < 64 * (EMBED / 4); idx += 256) {
    const int r = idx >> 7;             // / (EMBED/4)
    const int c4 = (idx & 127) << 2;
    const int grow = strip + r;
    const int b = grow / PADLEN;
    const int p = grow - b * PADLEN;
    const int sr = p - WIN;
    float4 v = make_float4(0.f, 0.f, 0.f, 0.f);
    if (sr >= 0 && sr < SEQ)
      v = *(const float4*)(x + ((size_t)b * SEQ + sr) * EMBED + c4);
    __bf16 h, l;
    __bf16* dh = As_hi + r * LDA + c4;
    __bf16* dl = As_lo + r * LDA + c4;
    cvt_split(v.x, h, l); dh[0] = h; dl[0] = l;
    cvt_split(v.y, h, l); dh[1] = h; dl[1] = l;
    cvt_split(v.z, h, l); dh[2] = h; dl[2] = l;
    cvt_split(v.w, h, l); dh[3] = h; dl[3] = l;
  }
  __syncthreads();

  const int wave = tid >> 5;
  const int lane = tid & 31;

  // 4 M-subtiles x 32 N-tiles = 128 wave tasks; each computes Q,K,V 16x16 tiles
  for (int task = wave; task < 128; task += 8) {
    const int msub = task & 3;
    const int n0 = (task >> 2) << 4;

    // Warm the weight rows this task will stream (global_prefetch_b8).
    {
      const size_t roff = (size_t)(n0 + (lane & 15)) * EMBED;
      __builtin_prefetch(Wqh + roff, 0, 1);
      __builtin_prefetch(Wql + roff, 0, 1);
      __builtin_prefetch(Wkh + roff, 0, 1);
      __builtin_prefetch(Wkl + roff, 0, 1);
      __builtin_prefetch(Wvh + roff, 0, 1);
      __builtin_prefetch(Wvl + roff, 0, 1);
    }

    v8f aq, ak, av;
#pragma unroll
    for (int i = 0; i < 8; ++i) { aq[i] = 0.f; ak[i] = 0.f; av[i] = 0.f; }

    for (int kb = 0; kb < EMBED; kb += 32) {
      const v16bf ah = load_afrag(As_hi, msub * 16, kb, lane);
      const v16bf al = load_afrag(As_lo, msub * 16, kb, lane);

      v16bf bh = load_bfrag(Wqh, n0, kb, lane);
      v16bf bl = load_bfrag(Wql, n0, kb, lane);
      aq = wmma_bf16(ah, bh, aq); aq = wmma_bf16(ah, bl, aq); aq = wmma_bf16(al, bh, aq);

      bh = load_bfrag(Wkh, n0, kb, lane);
      bl = load_bfrag(Wkl, n0, kb, lane);
      ak = wmma_bf16(ah, bh, ak); ak = wmma_bf16(ah, bl, ak); ak = wmma_bf16(al, bh, ak);

      bh = load_bfrag(Wvh, n0, kb, lane);
      bl = load_bfrag(Wvl, n0, kb, lane);
      av = wmma_bf16(ah, bh, av); av = wmma_bf16(ah, bl, av); av = wmma_bf16(al, bh, av);
    }

    // C/D layout: VGPR i -> M = (lane>=16 ? 8 : 0) + i, N = lane&15
    const int n = n0 + (lane & 15);
    const float bqv = bq[n], bkv = bk[n], bvv = bv[n];
    const int rbase = msub * 16 + ((lane >> 4) << 3);
#pragma unroll
    for (int i = 0; i < 8; ++i) {
      float prt = (aq[i] + bqv) * (ak[i] + bkv);
      prt += __shfl_xor(prt, 1);
      prt += __shfl_xor(prt, 2);
      prt += __shfl_xor(prt, 4);
      prt += __shfl_xor(prt, 8);          // reduced over the 16 columns of this tile
      if ((lane & 15) == 0) atomicAdd(&eSh[rbase + i], prt);   // ds_add_f32
      Vp[(size_t)(strip + rbase + i) * EMBED + n] = av[i] + bvv;
    }
  }
  __syncthreads();
  if (tid < 64) eOut[strip + tid] = eSh[tid] * 0.044194173824159216f; // 1/sqrt(512)
}

// ------- Kernel 2: windowed softmax + weighted V sum, LDS-tiled over 32 queries ---------
// Block = 256 thr (8 waves) handles TS=32 consecutive s of one batch. The 64 Vp rows
// covering all 32 windows are staged into LDS once (async global->LDS when available),
// cutting cache traffic 16x vs per-query streaming. Each wave reduces 4 queries from LDS.
__global__ __launch_bounds__(256) void attn_ctx_kernel(
    const float* __restrict__ eArr, const float* __restrict__ Vp,
    float* __restrict__ ctx) {
  extern __shared__ __bf16 smem[];
  float* Vs = (float*)smem;              // [ROWS_SH][EMBED] = 128 KB

  const int tid = threadIdx.x;
  const int blocksPerBatch = SEQ / TS;   // 64
  const int b = blockIdx.x / blocksPerBatch;
  const int sB = (blockIdx.x % blocksPerBatch) * TS;

  // Stage Vp rows [sB, sB+ROWS_SH) (padded coords) into LDS.
  const float* Vbase = Vp + (size_t)(b * PADLEN + sB) * EMBED;
  for (int idx = tid; idx < ROWS_SH * (EMBED / 4); idx += 256) {
    const int r = idx >> 7;
    const int c4 = (idx & 127) << 2;
    copy16_to_lds(Vbase + (size_t)r * EMBED + c4, Vs + r * EMBED + c4);
  }
  copy_join();                            // s_wait_asynccnt 0 (if async) + barrier

  const int wave = tid >> 5;
  const int lane = tid & 31;

#pragma unroll
  for (int i = 0; i < 4; ++i) {
    const int sl = wave * 4 + i;          // block-local query index, 0..31
    const float* ew = eArr + b * PADLEN + sB + sl;

    float mx = -3.0e38f;
#pragma unroll
    for (int w = 0; w < WINDOW; ++w) mx = fmaxf(mx, ew[w]);
    float den = 0.f;
#pragma unroll
    for (int w = 0; w < WINDOW; ++w) den += __expf(ew[w] - mx);
    const float rden = 1.0f / den;

    float acc[16];
#pragma unroll
    for (int j = 0; j < 16; ++j) acc[j] = 0.f;

    const float* Vrow = Vs + sl * EMBED + lane * 16;
    for (int w = 0; w < WINDOW; ++w) {
      const float aw = __expf(ew[w] - mx) * rden;
      const float4* v4 = (const float4*)(Vrow + w * EMBED);
#pragma unroll
      for (int j = 0; j < 4; ++j) {
        float4 v = v4[j];
        acc[4 * j + 0] = fmaf(aw, v.x, acc[4 * j + 0]);
        acc[4 * j + 1] = fmaf(aw, v.y, acc[4 * j + 1]);
        acc[4 * j + 2] = fmaf(aw, v.z, acc[4 * j + 2]);
        acc[4 * j + 3] = fmaf(aw, v.w, acc[4 * j + 3]);
      }
    }
    float4* o4 = (float4*)(ctx + (size_t)(b * SEQ + sB + sl) * EMBED + lane * 16);
#pragma unroll
    for (int j = 0; j < 4; ++j)
      o4[j] = make_float4(acc[4 * j], acc[4 * j + 1], acc[4 * j + 2], acc[4 * j + 3]);
  }
}

// ---------------- Kernel 3: out = ctx @ Wo.T + bo ---------------------------------------
__global__ __launch_bounds__(256) void out_proj_kernel(
    const float* __restrict__ ctx,
    const __bf16* __restrict__ Woh, const __bf16* __restrict__ Wol,
    const float* __restrict__ bo,
    float* __restrict__ out) {
  extern __shared__ __bf16 smem[];
  __bf16* As_hi = smem;
  __bf16* As_lo = smem + 64 * LDA;

  const int tid = threadIdx.x;
  const int strip = blockIdx.x * 64;

  for (int idx = tid; idx < 64 * (EMBED / 4); idx += 256) {
    const int r = idx >> 7;
    const int c4 = (idx & 127) << 2;
    float4 v = *(const float4*)(ctx + (size_t)(strip + r) * EMBED + c4);
    __bf16 h, l;
    __bf16* dh = As_hi + r * LDA + c4;
    __bf16* dl = As_lo + r * LDA + c4;
    cvt_split(v.x, h, l); dh[0] = h; dl[0] = l;
    cvt_split(v.y, h, l); dh[1] = h; dl[1] = l;
    cvt_split(v.z, h, l); dh[2] = h; dl[2] = l;
    cvt_split(v.w, h, l); dh[3] = h; dl[3] = l;
  }
  __syncthreads();

  const int wave = tid >> 5;
  const int lane = tid & 31;

  for (int task = wave; task < 128; task += 8) {
    const int msub = task & 3;
    const int n0 = (task >> 2) << 4;

    {
      const size_t roff = (size_t)(n0 + (lane & 15)) * EMBED;
      __builtin_prefetch(Woh + roff, 0, 1);
      __builtin_prefetch(Wol + roff, 0, 1);
    }

    v8f acc;
#pragma unroll
    for (int i = 0; i < 8; ++i) acc[i] = 0.f;

    for (int kb = 0; kb < EMBED; kb += 32) {
      const v16bf ah = load_afrag(As_hi, msub * 16, kb, lane);
      const v16bf al = load_afrag(As_lo, msub * 16, kb, lane);
      const v16bf bh = load_bfrag(Woh, n0, kb, lane);
      const v16bf bl = load_bfrag(Wol, n0, kb, lane);
      acc = wmma_bf16(ah, bh, acc);
      acc = wmma_bf16(ah, bl, acc);
      acc = wmma_bf16(al, bh, acc);
    }

    const int n = n0 + (lane & 15);
    const float bov = bo[n];
    const int rbase = msub * 16 + ((lane >> 4) << 3);
#pragma unroll
    for (int i = 0; i < 8; ++i)
      out[(size_t)(strip + rbase + i) * EMBED + n] = acc[i] + bov;
  }
}

extern "C" void kernel_launch(void* const* d_in, const int* in_sizes, int n_in,
                              void* d_out, int out_size, void* d_ws, size_t ws_size,
                              hipStream_t stream) {
  (void)in_sizes; (void)n_in; (void)out_size; (void)ws_size;

  const float* x  = (const float*)d_in[0];
  const float* Wq = (const float*)d_in[1];
  const float* bq = (const float*)d_in[2];
  const float* Wk = (const float*)d_in[3];
  const float* bk = (const float*)d_in[4];
  const float* Wv = (const float*)d_in[5];
  const float* bv = (const float*)d_in[6];
  const float* Wo = (const float*)d_in[7];
  const float* bo = (const float*)d_in[8];
  float* out = (float*)d_out;

  // Workspace: e | Vp | ctx | 8 bf16 weight planes (hi/lo x {q,k,v,o})
  char* ws = (char*)d_ws;
  size_t off = 0;
  float* eArr = (float*)(ws + off); off += ((size_t)ROWS_TOTAL * sizeof(float) + 255) & ~(size_t)255;
  float* Vp   = (float*)(ws + off); off += (size_t)ROWS_TOTAL * EMBED * sizeof(float);
  float* ctx  = (float*)(ws + off); off += (size_t)OUT_ROWS * EMBED * sizeof(float);
  const size_t wplane = (size_t)EMBED * EMBED * sizeof(__bf16);   // 512 KB
  __bf16* Wqh = (__bf16*)(ws + off); off += wplane;
  __bf16* Wql = (__bf16*)(ws + off); off += wplane;
  __bf16* Wkh = (__bf16*)(ws + off); off += wplane;
  __bf16* Wkl = (__bf16*)(ws + off); off += wplane;
  __bf16* Wvh = (__bf16*)(ws + off); off += wplane;
  __bf16* Wvl = (__bf16*)(ws + off); off += wplane;
  __bf16* Woh = (__bf16*)(ws + off); off += wplane;
  __bf16* Wol = (__bf16*)(ws + off); off += wplane;

  const size_t ldsGemm = (size_t)2 * 64 * LDA * sizeof(__bf16);   // 133,120 B
  const size_t ldsAttn = (size_t)ROWS_SH * EMBED * sizeof(float); // 131,072 B
  (void)hipFuncSetAttribute((const void*)qkv_e_v_kernel,
                            hipFuncAttributeMaxDynamicSharedMemorySize, (int)ldsGemm);
  (void)hipFuncSetAttribute((const void*)attn_ctx_kernel,
                            hipFuncAttributeMaxDynamicSharedMemorySize, (int)ldsAttn);
  (void)hipFuncSetAttribute((const void*)out_proj_kernel,
                            hipFuncAttributeMaxDynamicSharedMemorySize, (int)ldsGemm);

  // One-time weight split (re-done every call for determinism; 1 M elems, negligible)
  const int wsplit_blocks = (EMBED * EMBED) / (256 * 4);   // 256
  wsplit_kernel<<<wsplit_blocks, 256, 0, stream>>>(Wq, Wqh, Wql);
  wsplit_kernel<<<wsplit_blocks, 256, 0, stream>>>(Wk, Wkh, Wkl);
  wsplit_kernel<<<wsplit_blocks, 256, 0, stream>>>(Wv, Wvh, Wvl);
  wsplit_kernel<<<wsplit_blocks, 256, 0, stream>>>(Wo, Woh, Wol);

  qkv_e_v_kernel<<<ROWS_TOTAL / 64, 256, ldsGemm, stream>>>(
      x, Wqh, Wql, bq, Wkh, Wkl, bk, Wvh, Wvl, bv, eArr, Vp);
  attn_ctx_kernel<<<OUT_ROWS / TS, 256, ldsAttn, stream>>>(eArr, Vp, ctx);
  out_proj_kernel<<<OUT_ROWS / 64, 256, ldsGemm, stream>>>(ctx, Woh, Wol, bo, out);
}
